// Net_32469952758301
// MI455X (gfx1250) — matrix operand order repaired and checked by reading, not compiled, not run on hardware
//
#include <hip/hip_runtime.h>
#include <hip/hip_bf16.h>
#include <math.h>

// ---------------- problem constants (match reference) ----------------
constexpr int NN    = 50000;          // nodes
constexpr int BB    = 256;            // graphs
constexpr int FIN   = 128;
constexpr int DD    = 384;            // 3*H
constexpr int CC    = 10;
constexpr int STEPS = 10;
constexpr int KQH   = 1152;           // 2*D + D  (LSTM concat K)
constexpr int GG    = 1536;           // 4*D gates

typedef __attribute__((ext_vector_type(16))) __bf16 v16bf;
typedef __attribute__((ext_vector_type(8)))  __bf16 v8bf;
typedef __attribute__((ext_vector_type(8)))  float  v8f;

// ---------------- WMMA helpers ----------------
__device__ inline v8f wmma_bf16f32(v16bf a, v16bf b, v8f c) {
  return __builtin_amdgcn_wmma_f32_16x16x32_bf16(false, a, false, b, (short)0, c, false, false);
}

// A fragment (16x32 bf16) from bf16 row-major [rows, ld].
// ISA layout: lane row = lane%16, half h = lane/16;
// fragment elem i(0..7)  <-> K = k0 + 8h + i        (contiguous 16B)
// fragment elem i(8..15) <-> K = k0 + 16 + 8h + (i-8) (contiguous 16B)
__device__ inline v16bf load_A_frag_bf16(const __bf16* A, int ld, int m0, int k0, int lane) {
  int r = m0 + (lane & 15);
  int h = (lane >> 4) & 1;
  const __bf16* row = A + (size_t)r * ld + k0 + 8 * h;
  v8bf lo = *(const v8bf*)(row);
  v8bf hi = *(const v8bf*)(row + 16);
  v16bf a;
#pragma unroll
  for (int i = 0; i < 8; ++i) { a[i] = lo[i]; a[8 + i] = hi[i]; }
  return a;
}

// B fragment (32x16 bf16) from transposed bf16 weights Wt[Ncols][K] (ld = K).
// ISA layout: lane col = lane%16; elem i(0..7) <-> K = k0 + 8h + i, elem i(8..15) <-> K = k0 + 16 + 8h + (i-8).
__device__ inline v16bf load_B_frag_bf16t(const __bf16* Wt, int ld, int k0, int n0, int lane) {
  int n = n0 + (lane & 15);
  int h = (lane >> 4) & 1;
  const __bf16* col = Wt + (size_t)n * ld + k0 + 8 * h;
  v8bf lo = *(const v8bf*)(col);
  v8bf hi = *(const v8bf*)(col + 16);
  v16bf b;
#pragma unroll
  for (int i = 0; i < 8; ++i) { b[i] = lo[i]; b[8 + i] = hi[i]; }
  return b;
}

__device__ inline void atomic_add_f32(float* p, float v) {
  __hip_atomic_fetch_add(p, v, __ATOMIC_RELAXED, __HIP_MEMORY_SCOPE_AGENT);
}

__device__ inline float sigmoidf(float x) { return 1.0f / (1.0f + __expf(-x)); }

// ---------------- prep kernels (run once, cheap) ----------------

__global__ __launch_bounds__(256) void cast_f32_to_bf16(const float* __restrict__ s,
                                                        __bf16* __restrict__ d, int n) {
  int i = blockIdx.x * 256 + threadIdx.x;
  if (i < n) d[i] = (__bf16)s[i];
}

// Wt[n][k] = bf16(W[k][n]) for 128x128
__global__ __launch_bounds__(256) void transpose_cast_128(const float* __restrict__ W,
                                                          __bf16* __restrict__ Wt) {
  int t = blockIdx.x * 256 + threadIdx.x;   // < 16384
  int n = t >> 7, k = t & 127;
  Wt[(size_t)n * 128 + k] = (__bf16)W[(size_t)k * 128 + n];
}

// Wcat[g][k] = bf16( k<768 ? W_ih[g][k] : W_hh[g][k-768] )  -> [1536,1152]
__global__ __launch_bounds__(256) void build_wcat(const float* __restrict__ Wih,
                                                  const float* __restrict__ Whh,
                                                  __bf16* __restrict__ Wcat) {
  int t = blockIdx.x * 256 + threadIdx.x;
  if (t >= GG * KQH) return;
  int g = t / KQH, k = t - g * KQH;
  float v = (k < 2 * DD) ? Wih[(size_t)g * (2 * DD) + k] : Whh[(size_t)g * DD + (k - 2 * DD)];
  Wcat[(size_t)g * KQH + k] = (__bf16)v;
}

// ---------------- graph kernels ----------------

// agg[dst[e], f] += (float)Xb[src[e], f]   (bf16 source, f32 atomic accumulate)
__global__ __launch_bounds__(256) void edge_aggregate(const __bf16* __restrict__ Xb, int ldx,
                                                      const int* __restrict__ src,
                                                      const int* __restrict__ dst,
                                                      float* __restrict__ agg, int nE) {
  long long t = (long long)blockIdx.x * blockDim.x + threadIdx.x;
  if (t >= (long long)nE * 128) return;
  int e = (int)(t >> 7);
  int f = (int)(t & 127);
  int s = src[e];
  int d = dst[e];
  atomic_add_f32(&agg[(size_t)d * 128 + f], (float)Xb[(size_t)s * ldx + f]);
}

// Y/Yb[:, ycol:ycol+128] = relu(Xb @ Wroot + Aggb @ Wrel + bias), bf16 WMMA, f32 accum.
// One block = one 16-row M-tile; 8 waves cover the 128 output columns.
__global__ __launch_bounds__(256) void graphconv_gemm(const __bf16* __restrict__ Xb, int ldx,
                                                      const __bf16* __restrict__ Aggb,
                                                      const __bf16* __restrict__ WrootT,
                                                      const __bf16* __restrict__ WrelT,
                                                      const float* __restrict__ bias,
                                                      float* __restrict__ Y,
                                                      __bf16* __restrict__ Yb,
                                                      int ldy, int ycol) {
  int lane = threadIdx.x & 31;
  int wid  = threadIdx.x >> 5;          // 0..7
  int m0   = blockIdx.x * 16;
  int n0   = wid * 16;
  v8f acc = {};
#pragma unroll
  for (int k0 = 0; k0 < 128; k0 += 32) {
    v16bf a1 = load_A_frag_bf16(Xb, ldx, m0, k0, lane);
    v16bf b1 = load_B_frag_bf16t(WrootT, 128, k0, n0, lane);
    acc = wmma_bf16f32(a1, b1, acc);
    v16bf a2 = load_A_frag_bf16(Aggb, 128, m0, k0, lane);
    v16bf b2 = load_B_frag_bf16t(WrelT, 128, k0, n0, lane);
    acc = wmma_bf16f32(a2, b2, acc);
  }
  int col = n0 + (lane & 15);
  int hh  = (lane >> 4) & 1;
  float bv = bias[col];
#pragma unroll
  for (int r2 = 0; r2 < 8; ++r2) {
    int m = m0 + r2 + 8 * hh;
    if (m < NN) {
      float v = acc[r2] + bv;
      v = v > 0.f ? v : 0.f;
      Y [(size_t)m * ldy + ycol + col] = v;
      Yb[(size_t)m * ldy + ycol + col] = (__bf16)v;
    }
  }
}

// segment start offsets from sorted batch array; starts[BB] = NN
__global__ __launch_bounds__(256) void seg_bounds(const int* __restrict__ batch,
                                                  int* __restrict__ starts, int n) {
  int i = blockIdx.x * blockDim.x + threadIdx.x;
  if (i >= n) return;
  int b = batch[i];
  if (i == 0) {
    for (int g = 0; g <= b; ++g) starts[g] = 0;
  } else {
    int pb = batch[i - 1];
    for (int g = pb + 1; g <= b; ++g) starts[g] = i;
  }
  if (i == n - 1) {
    for (int g = b + 1; g <= BB; ++g) starts[g] = n;
  }
}

// ---------------- Set2Set kernels ----------------

// gates[256,1536] = qh @ Wcat^T + (b_ih + b_hh)   (qh bf16 [256,1152], Wcat bf16 [1536,1152])
__global__ __launch_bounds__(256) void lstm_gates_gemm(const __bf16* __restrict__ qh,
                                                       const __bf16* __restrict__ Wcat,
                                                       const float* __restrict__ b_ih,
                                                       const float* __restrict__ b_hh,
                                                       float* __restrict__ gates) {
  int lane = threadIdx.x & 31;
  int gt   = blockIdx.x * 8 + (threadIdx.x >> 5);   // 0..95
  int m0   = blockIdx.y * 16;                       // 0..255
  int g0   = gt * 16;
  v8f acc = {};
  for (int k0 = 0; k0 < KQH; k0 += 32) {
    v16bf a = load_A_frag_bf16(qh, KQH, m0, k0, lane);
    v16bf b = load_B_frag_bf16t(Wcat, KQH, k0, g0, lane);
    acc = wmma_bf16f32(a, b, acc);
  }
  int col = g0 + (lane & 15);
  int hh  = (lane >> 4) & 1;
  float bv = b_ih[col] + b_hh[col];
#pragma unroll
  for (int r2 = 0; r2 < 8; ++r2) {
    int m = m0 + r2 + 8 * hh;
    gates[(size_t)m * GG + col] = acc[r2] + bv;
  }
}

// c = f*c + i*tanh(g); h = o*tanh(c); update f32 h/c/qstar-q and bf16 qh (q and h slots)
__global__ __launch_bounds__(256) void lstm_pointwise(const float* __restrict__ gates,
                                                      float* __restrict__ c,
                                                      float* __restrict__ h,
                                                      float* __restrict__ qstar,
                                                      __bf16* __restrict__ qh) {
  int t = blockIdx.x * blockDim.x + threadIdx.x;
  if (t >= BB * DD) return;
  int b = t / DD, d = t - b * DD;
  const float* g = gates + (size_t)b * GG;
  float ig = sigmoidf(g[d]);
  float fg = sigmoidf(g[DD + d]);
  float gg = tanhf(g[2 * DD + d]);
  float og = sigmoidf(g[3 * DD + d]);
  float cn = fg * c[t] + ig * gg;
  c[t] = cn;
  float hn = og * tanhf(cn);
  h[t] = hn;
  qstar[(size_t)b * (2 * DD) + d] = hn;                 // q part (f32, for MLP head)
  __bf16 hb = (__bf16)hn;
  qh[(size_t)b * KQH + d]          = hb;                // q part (bf16, for gate GEMM)
  qh[(size_t)b * KQH + 2 * DD + d] = hb;                // h part (bf16)
}

// e[n] = dot(xc[n,:], h[batch[n],:])   one wave per node
__global__ __launch_bounds__(256) void attn_dot(const float* __restrict__ xc,
                                                const float* __restrict__ h,
                                                const int* __restrict__ batch,
                                                float* __restrict__ ev, int n) {
  int node = blockIdx.x * 8 + (threadIdx.x >> 5);
  int lane = threadIdx.x & 31;
  if (node >= n) return;
  int b = batch[node];
  const float* xr = xc + (size_t)node * DD;
  const float* hr = h + (size_t)b * DD;
  float s = 0.f;
  for (int k = lane; k < DD; k += 32) s += xr[k] * hr[k];
#pragma unroll
  for (int off = 16; off > 0; off >>= 1) s += __shfl_xor(s, off, 32);
  if (lane == 0) ev[node] = s;
}

// per-graph softmax over e; r = sum a*xc; writes qstar[:,384:768] (f32) and qh r-slot (bf16)
__global__ __launch_bounds__(256) void attn_softmax_r(const float* __restrict__ xc,
                                                      const float* __restrict__ ev,
                                                      const int* __restrict__ starts,
                                                      float* __restrict__ qstar,
                                                      __bf16* __restrict__ qh) {
  __shared__ float red[256];
  __shared__ float av[256];
  int b = blockIdx.x;
  int s = starts[b], e = starts[b + 1];
  int tid = threadIdx.x;
  // max
  float lm = -3.4e38f;
  for (int i = s + tid; i < e; i += 256) lm = fmaxf(lm, ev[i]);
  red[tid] = lm; __syncthreads();
  for (int off = 128; off > 0; off >>= 1) {
    if (tid < off) red[tid] = fmaxf(red[tid], red[tid + off]);
    __syncthreads();
  }
  float m = red[0]; __syncthreads();
  // sum of exp
  float ls = 0.f;
  for (int i = s + tid; i < e; i += 256) ls += __expf(ev[i] - m);
  red[tid] = ls; __syncthreads();
  for (int off = 128; off > 0; off >>= 1) {
    if (tid < off) red[tid] += red[tid + off];
    __syncthreads();
  }
  float inv = (red[0] > 0.f) ? 1.0f / red[0] : 0.f;
  __syncthreads();
  // r accumulation: thread covers dims {tid} and {tid+256 if < 384}
  float acc0 = 0.f, acc1 = 0.f;
  for (int base = s; base < e; base += 256) {
    int cnt = min(256, e - base);
    __syncthreads();
    if (tid < cnt) av[tid] = __expf(ev[base + tid] - m) * inv;
    __syncthreads();
    for (int i = 0; i < cnt; ++i) {
      float a = av[i];
      const float* xr = xc + (size_t)(base + i) * DD;
      acc0 += a * xr[tid];
      if (tid < DD - 256) acc1 += a * xr[tid + 256];
    }
  }
  qstar[(size_t)b * (2 * DD) + DD + tid] = acc0;
  qh[(size_t)b * KQH + DD + tid] = (__bf16)acc0;
  if (tid < DD - 256) {
    qstar[(size_t)b * (2 * DD) + DD + tid + 256] = acc1;
    qh[(size_t)b * KQH + DD + tid + 256] = (__bf16)acc1;
  }
}

// MLP head + log_softmax; one block (128 threads) per graph
__global__ __launch_bounds__(128) void mlp_head(const float* __restrict__ p,     // [256,768]
                                                const float* __restrict__ l1W, const float* __restrict__ l1b,
                                                const float* __restrict__ l2W, const float* __restrict__ l2b,
                                                const float* __restrict__ l3W, const float* __restrict__ l3b,
                                                float* __restrict__ out) {
  __shared__ float h1[128];
  __shared__ float h2[64];
  __shared__ float lg[10];
  int b = blockIdx.x, tid = threadIdx.x;
  const float* pr = p + (size_t)b * (2 * DD);
  float s = l1b[tid];
  for (int k = 0; k < 2 * DD; ++k) s += pr[k] * l1W[(size_t)k * 128 + tid];
  h1[tid] = fmaxf(s, 0.f);
  __syncthreads();
  if (tid < 64) {
    float s2 = l2b[tid];
    for (int k = 0; k < 128; ++k) s2 += h1[k] * l2W[(size_t)k * 64 + tid];
    h2[tid] = fmaxf(s2, 0.f);
  }
  __syncthreads();
  if (tid < CC) {
    float s3 = l3b[tid];
    for (int k = 0; k < 64; ++k) s3 += h2[k] * l3W[(size_t)k * CC + tid];
    lg[tid] = s3;
  }
  __syncthreads();
  if (tid == 0) {
    float m = lg[0];
    for (int j = 1; j < CC; ++j) m = fmaxf(m, lg[j]);
    float den = 0.f;
    for (int j = 0; j < CC; ++j) den += __expf(lg[j] - m);
    float lse = m + __logf(den);
    for (int j = 0; j < CC; ++j) out[(size_t)b * CC + j] = lg[j] - lse;
  }
}

// ---------------- host launch ----------------
static inline size_t align256(size_t x) { return (x + 255) & ~(size_t)255; }

extern "C" void kernel_launch(void* const* d_in, const int* in_sizes, int n_in,
                              void* d_out, int out_size, void* d_ws, size_t ws_size,
                              hipStream_t stream) {
  (void)n_in; (void)out_size; (void)ws_size;
  const float* x        = (const float*)d_in[0];
  const int*   eidx     = (const int*)d_in[1];
  const int*   batch    = (const int*)d_in[2];
  const float* W1_root  = (const float*)d_in[3];
  const float* W1_rel   = (const float*)d_in[4];
  const float* b1       = (const float*)d_in[5];
  const float* W2_root  = (const float*)d_in[6];
  const float* W2_rel   = (const float*)d_in[7];
  const float* b2       = (const float*)d_in[8];
  const float* W3_root  = (const float*)d_in[9];
  const float* W3_rel   = (const float*)d_in[10];
  const float* b3       = (const float*)d_in[11];
  const float* W_ih     = (const float*)d_in[12];
  const float* W_hh     = (const float*)d_in[13];
  const float* b_ih     = (const float*)d_in[14];
  const float* b_hh     = (const float*)d_in[15];
  const float* lin1_W   = (const float*)d_in[16];
  const float* lin1_b   = (const float*)d_in[17];
  const float* lin2_W   = (const float*)d_in[18];
  const float* lin2_b   = (const float*)d_in[19];
  const float* lin3_W   = (const float*)d_in[20];
  const float* lin3_b   = (const float*)d_in[21];

  const int nE = in_sizes[1] / 2;
  const int*  src = eidx;
  const int*  dst = eidx + nE;
  float* out = (float*)d_out;

  // workspace layout
  char* base = (char*)d_ws;
  size_t off = 0;
  float*  xc    = (float*)(base + off);  off = align256(off + (size_t)NN * DD * 4);      // f32 [N,384]
  __bf16* xcb   = (__bf16*)(base + off); off = align256(off + (size_t)NN * DD * 2);      // bf16 [N,384]
  __bf16* xb    = (__bf16*)(base + off); off = align256(off + (size_t)NN * 128 * 2);     // bf16 [N,128] input copy
  float*  agg   = (float*)(base + off);  off = align256(off + (size_t)NN * 128 * 4);     // f32 [N,128]
  __bf16* aggb  = (__bf16*)(base + off); off = align256(off + (size_t)NN * 128 * 2);     // bf16 [N,128]
  float*  ev    = (float*)(base + off);  off = align256(off + (size_t)NN * 4);           // f32 [N]
  float*  gates = (float*)(base + off);  off = align256(off + (size_t)BB * GG * 4);      // f32 [256,1536]
  float*  qstar = (float*)(base + off);  off += (size_t)BB * (2 * DD) * 4;               // f32 [256,768]
  float*  hbuf  = (float*)(base + off);  off += (size_t)BB * DD * 4;                     // f32 [256,384]
  float*  cbuf  = (float*)(base + off);  off += (size_t)BB * DD * 4;                     // f32 [256,384]
  off = align256(off);
  __bf16* qh    = (__bf16*)(base + off); off = align256(off + (size_t)BB * KQH * 2);     // bf16 [256,1152]
  __bf16* Wt[6];
  for (int i = 0; i < 6; ++i) { Wt[i] = (__bf16*)(base + off); off = align256(off + 128 * 128 * 2); }
  __bf16* Wcat  = (__bf16*)(base + off); off = align256(off + (size_t)GG * KQH * 2);     // bf16 [1536,1152]
  int*    starts = (int*)(base + off);   off = align256(off + (size_t)(BB + 1) * 4);

  const dim3 blk256(256), blk128(128);
  const int edgeBlocks = (int)(((long long)nE * 128 + 255) / 256);
  const int gemmBlocks = NN / 16;
  const int castBlocks = (NN * 128 + 255) / 256;

  // ---- one-time prep: bf16 input copy, transposed bf16 weights, concat LSTM weights ----
  cast_f32_to_bf16<<<castBlocks, blk256, 0, stream>>>(x, xb, NN * 128);
  transpose_cast_128<<<64, blk256, 0, stream>>>(W1_root, Wt[0]);
  transpose_cast_128<<<64, blk256, 0, stream>>>(W1_rel,  Wt[1]);
  transpose_cast_128<<<64, blk256, 0, stream>>>(W2_root, Wt[2]);
  transpose_cast_128<<<64, blk256, 0, stream>>>(W2_rel,  Wt[3]);
  transpose_cast_128<<<64, blk256, 0, stream>>>(W3_root, Wt[4]);
  transpose_cast_128<<<64, blk256, 0, stream>>>(W3_rel,  Wt[5]);
  build_wcat<<<(GG * KQH + 255) / 256, blk256, 0, stream>>>(W_ih, W_hh, Wcat);

  // ---- GraphConv layer 1 ----
  hipMemsetAsync(agg, 0, (size_t)NN * 128 * 4, stream);
  edge_aggregate<<<edgeBlocks, blk256, 0, stream>>>(xb, 128, src, dst, agg, nE);
  cast_f32_to_bf16<<<castBlocks, blk256, 0, stream>>>(agg, aggb, NN * 128);
  graphconv_gemm<<<gemmBlocks, blk256, 0, stream>>>(xb, 128, aggb, Wt[0], Wt[1], b1, xc, xcb, DD, 0);
  // ---- layer 2 (input = cols 0..127) ----
  hipMemsetAsync(agg, 0, (size_t)NN * 128 * 4, stream);
  edge_aggregate<<<edgeBlocks, blk256, 0, stream>>>(xcb, DD, src, dst, agg, nE);
  cast_f32_to_bf16<<<castBlocks, blk256, 0, stream>>>(agg, aggb, NN * 128);
  graphconv_gemm<<<gemmBlocks, blk256, 0, stream>>>(xcb, DD, aggb, Wt[2], Wt[3], b2, xc, xcb, DD, 128);
  // ---- layer 3 (input = cols 128..255) ----
  hipMemsetAsync(agg, 0, (size_t)NN * 128 * 4, stream);
  edge_aggregate<<<edgeBlocks, blk256, 0, stream>>>(xcb + 128, DD, src, dst, agg, nE);
  cast_f32_to_bf16<<<castBlocks, blk256, 0, stream>>>(agg, aggb, NN * 128);
  graphconv_gemm<<<gemmBlocks, blk256, 0, stream>>>(xcb + 128, DD, aggb, Wt[4], Wt[5], b3, xc, xcb, DD, 256);

  // ---- segment boundaries + Set2Set state init ----
  seg_bounds<<<(NN + 255) / 256, blk256, 0, stream>>>(batch, starts, NN);
  hipMemsetAsync(qstar, 0, (size_t)BB * (2 * DD + DD + DD) * 4, stream);  // qstar, h, c contiguous
  hipMemsetAsync(qh, 0, (size_t)BB * KQH * 2, stream);

  // ---- Set2Set: 10 steps ----
  for (int st = 0; st < STEPS; ++st) {
    lstm_gates_gemm<<<dim3(12, 16), blk256, 0, stream>>>(qh, Wcat, b_ih, b_hh, gates);
    lstm_pointwise<<<(BB * DD + 255) / 256, blk256, 0, stream>>>(gates, cbuf, hbuf, qstar, qh);
    attn_dot<<<(NN + 7) / 8, blk256, 0, stream>>>(xc, hbuf, batch, ev, NN);
    attn_softmax_r<<<BB, blk256, 0, stream>>>(xc, ev, starts, qstar, qh);
  }

  // ---- MLP head ----
  mlp_head<<<BB, blk128, 0, stream>>>(qstar, lin1_W, lin1_b, lin2_W, lin2_b, lin3_W, lin3_b, out);
}
